// LSTM_26972394619446
// MI455X (gfx1250) — compile-verified, matrix operand order checked by reading, not run on hardware
//
#include <hip/hip_runtime.h>
#include <hip/hip_bf16.h>
#include <stdint.h>

// ---------------------------------------------------------------------------
// LSTM forward for MI455X (gfx1250, wave32, WMMA).
//   B=64, T=512, I=256, H=1024, O=256
// v2: recurrent weights staged in LDS (128 KB/block, fragment-order layout),
//     dual accumulators to halve the WMMA RAW chain, x-row prefetch.
// ---------------------------------------------------------------------------

typedef __attribute__((ext_vector_type(16))) __bf16 v16bf;
typedef __attribute__((ext_vector_type(8)))  float  v8f;

constexpr int kB = 64, kT = 512, kI = 256, kH = 1024, kO = 256;

// ---------------- fragment load (A or B operand), bf16 16x32 ---------------
// Lane (l%16) selects the row; lanes 0-15 hold K={k0..k0+7, k0+16..k0+23},
// lanes 16-31 hold K={k0+8..k0+15, k0+24..k0+31}  (ISA 7.12.2 16-bit A layout).
__device__ inline v16bf ld_frag(const __hip_bfloat16* __restrict__ row, int k0, int hiOff) {
    union { v16bf v; uint4 u[2]; } f;
    const __hip_bfloat16* p = row + k0 + hiOff;
    f.u[0] = *reinterpret_cast<const uint4*>(p);
    f.u[1] = *reinterpret_cast<const uint4*>(p + 16);
    return f.v;
}

__device__ inline v16bf ld_frag_lds(const uint4* p) {
    union { v16bf v; uint4 u[2]; } f;
    f.u[0] = p[0];
    f.u[1] = p[1];
    return f.v;
}

__device__ inline v8f wmma_bf16(v16bf a, v16bf b, v8f c) {
    // (neg_a, A, neg_b, B, c_mod, C, reuse_a, reuse_b)
    return __builtin_amdgcn_wmma_f32_16x16x32_bf16(false, a, false, b, (short)0, c, false, false);
}

__device__ inline float sigm_f(float x) { return 1.0f / (1.0f + __expf(-x)); }
__device__ inline float tanh_f(float x) {
    float ax = fabsf(x);
    float e  = __expf(-2.0f * ax);
    float t  = (1.0f - e) / (1.0f + e);
    return copysignf(t, x);
}

// ---------------- device-wide barrier (persistent kernel) ------------------
__device__ inline void grid_barrier(unsigned* bar, unsigned nblk) {
    __threadfence();
    __syncthreads();
    if (threadIdx.x == 0) {
        unsigned gen = __hip_atomic_load(&bar[1], __ATOMIC_ACQUIRE, __HIP_MEMORY_SCOPE_AGENT);
        unsigned tic = __hip_atomic_fetch_add(&bar[0], 1u, __ATOMIC_ACQ_REL, __HIP_MEMORY_SCOPE_AGENT);
        if (tic == nblk - 1u) {
            __hip_atomic_store(&bar[0], 0u, __ATOMIC_RELAXED, __HIP_MEMORY_SCOPE_AGENT);
            __hip_atomic_fetch_add(&bar[1], 1u, __ATOMIC_ACQ_REL, __HIP_MEMORY_SCOPE_AGENT);
        } else {
            while (__hip_atomic_load(&bar[1], __ATOMIC_ACQUIRE, __HIP_MEMORY_SCOPE_AGENT) == gen)
                __builtin_amdgcn_s_sleep(2);
        }
    }
    __syncthreads();
}

// ---------------- small utility kernels ------------------------------------
__global__ void f32_to_bf16_kernel(const float* __restrict__ src,
                                   __hip_bfloat16* __restrict__ dst, int n) {
    int i = blockIdx.x * blockDim.x + threadIdx.x;
    int stride = gridDim.x * blockDim.x;
    for (; i < n; i += stride) dst[i] = __float2bfloat16(src[i]);
}

__global__ void bar_init_kernel(unsigned* bar) {
    if (threadIdx.x == 0) { bar[0] = 0u; bar[1] = 0u; }
}

// ---------------- persistent recurrent scan --------------------------------
// grid = 256 blocks (one 16x16 output tile each), block = 128 (4 waves).
// wave g in {0:i, 1:o, 2:f, 3:z} computes its gate's preactivation tile.
// Wh slices live in LDS in WMMA-fragment order: [gate][ks][lane][2 x uint4].
__global__ __launch_bounds__(128) void lstm_scan_kernel(
    const __hip_bfloat16* __restrict__ xbf,
    const __hip_bfloat16* __restrict__ wx0, const __hip_bfloat16* __restrict__ wx1,
    const __hip_bfloat16* __restrict__ wx2, const __hip_bfloat16* __restrict__ wx3,
    const __hip_bfloat16* __restrict__ wh0, const __hip_bfloat16* __restrict__ wh1,
    const __hip_bfloat16* __restrict__ wh2, const __hip_bfloat16* __restrict__ wh3,
    const float* __restrict__ bias_i, const float* __restrict__ bias_o,
    const float* __restrict__ bias_f, const float* __restrict__ bias_z,
    const float* __restrict__ bh,
    __hip_bfloat16* hs_bf, float* cbuf, float* hs_f32, unsigned* bar)
{
    const int tid     = threadIdx.x;
    const int lane    = tid & 31;
    const int gate    = tid >> 5;        // 0..3
    const int tile    = blockIdx.x;      // 0..255
    const int mT      = tile >> 6;       // 0..3   (batch tiles, B=64)
    const int nT      = tile & 63;       // 0..63  (H tiles)
    const int laneLow = lane & 15;
    const int hi      = lane >> 4;
    const int hiOff   = hi * 8;          // K-chunk offset (elements)

    const int bRow = mT * 16 + laneLow;  // A row: batch index handled by lane
    const int jRow = nT * 16 + laneLow;  // B row: output column j (W row)

    const __hip_bfloat16* Wx = (gate == 0) ? wx0 : (gate == 1) ? wx1 : (gate == 2) ? wx2 : wx3;
    const __hip_bfloat16* Wh = (gate == 0) ? wh0 : (gate == 1) ? wh1 : (gate == 2) ? wh2 : wh3;
    const float*          bs = (gate == 0) ? bias_i : (gate == 1) ? bias_o
                              : (gate == 2) ? bias_f : bias_z;

    const __hip_bfloat16* wxRow = Wx + (size_t)jRow * kI;
    const __hip_bfloat16* whRow = Wh + (size_t)jRow * kH;
    const float bv = bs[nT * 16 + laneLow];   // C column = lane&15

    // 128 KB weight stage + 4 KB gate exchange (<= 320 KB/WGP per ISA)
    __shared__ uint4 wlds[4][32][32][2];
    __shared__ float glds[4][8][32];

    // ---- stage this block's Wh slice into LDS, fragment-order ----
#pragma unroll 4
    for (int ks = 0; ks < kH / 32; ++ks) {
        union { v16bf v; uint4 u[2]; } f;
        const __hip_bfloat16* p = whRow + ks * 32 + hiOff;
        f.u[0] = *reinterpret_cast<const uint4*>(p);
        f.u[1] = *reinterpret_cast<const uint4*>(p + 16);
        wlds[gate][ks][lane][0] = f.u[0];
        wlds[gate][ks][lane][1] = f.u[1];
    }
    __syncthreads();

    for (int t = 0; t < kT; ++t) {
        v8f acc0 = {bv, bv, bv, bv, bv, bv, bv, bv};
        v8f acc1 = {0.f, 0.f, 0.f, 0.f, 0.f, 0.f, 0.f, 0.f};

        // x_t projection: K = I = 256 (Wx fragments stay register-resident)
        const __hip_bfloat16* xRow = xbf + ((size_t)bRow * kT + t) * kI;
        if (t + 1 < kT) __builtin_prefetch(xRow + kI, 0, 1);   // next step's x row
#pragma unroll
        for (int ks = 0; ks < kI / 32; ks += 2) {
            acc0 = wmma_bf16(ld_frag(xRow, ks * 32, hiOff),
                             ld_frag(wxRow, ks * 32, hiOff), acc0);
            acc1 = wmma_bf16(ld_frag(xRow, (ks + 1) * 32, hiOff),
                             ld_frag(wxRow, (ks + 1) * 32, hiOff), acc1);
        }

        // h_{t-1} projection: K = H = 1024, B from LDS  (h_0 == 0 -> skip)
        if (t != 0) {
            const __hip_bfloat16* hRow = hs_bf + ((size_t)bRow * kT + (t - 1)) * kH;
#pragma unroll 4
            for (int ks = 0; ks < kH / 32; ks += 2) {
                acc0 = wmma_bf16(ld_frag(hRow, ks * 32, hiOff),
                                 ld_frag_lds(&wlds[gate][ks][lane][0]), acc0);
                acc1 = wmma_bf16(ld_frag(hRow, (ks + 1) * 32, hiOff),
                                 ld_frag_lds(&wlds[gate][ks + 1][lane][0]), acc1);
            }
        }

        v8f acc = acc0 + acc1;
#pragma unroll
        for (int r = 0; r < 8; ++r) glds[gate][r][lane] = acc[r];
        __syncthreads();

        if (gate == 0) {   // wave 0 fuses the elementwise cell update
            const int j = nT * 16 + laneLow;
#pragma unroll
            for (int r = 0; r < 8; ++r) {
                const int b = mT * 16 + r + hiOff;   // C row = r + 8*hi
                float gi = glds[0][r][lane], go = glds[1][r][lane];
                float gf = glds[2][r][lane], gz = glds[3][r][lane];
                float iv = sigm_f(gi), ov = sigm_f(go), fv = sigm_f(gf), zv = tanh_f(gz);
                size_t ci   = (size_t)b * kH + j;
                float  cold = (t == 0) ? 0.0f : cbuf[ci];
                float  cnew = __builtin_fmaf(fv, cold, iv * zv);
                cbuf[ci] = cnew;
                size_t hidx = ((size_t)b * kT + t) * kH + j;
                float  hv   = __builtin_fmaf(ov, tanh_f(cnew), bh[hidx]);
                hs_f32[hidx] = hv;
                hs_bf[hidx]  = __float2bfloat16(hv);
            }
        }
        grid_barrier(bar, gridDim.x);
    }
}

// ---------------- output projection y = hstore @ Wy^T + by -----------------
// M = B*T = 32768, N = O = 256, K = H = 1024. One wave per 16x16 tile.
__global__ __launch_bounds__(256) void out_proj_kernel(
    const __hip_bfloat16* __restrict__ hs_bf,
    const __hip_bfloat16* __restrict__ wy,
    const float* __restrict__ by, float* __restrict__ y)
{
    const int tid     = threadIdx.x;
    const int lane    = tid & 31;
    const int warp    = tid >> 5;
    const int gw      = blockIdx.x * 8 + warp;   // 0..32767
    const int mT      = gw >> 4;                 // 0..2047
    const int nT      = gw & 15;                 // 0..15
    const int laneLow = lane & 15;
    const int hiOff   = (lane >> 4) * 8;

    const int aRow = mT * 16 + laneLow;          // flattened (b*T + t)
    const int oCol = nT * 16 + laneLow;

    const __hip_bfloat16* hRow = hs_bf + (size_t)aRow * kH;
    const __hip_bfloat16* wRow = wy    + (size_t)oCol * kH;

    const float bv = by[oCol];
    v8f acc0 = {bv, bv, bv, bv, bv, bv, bv, bv};
    v8f acc1 = {0.f, 0.f, 0.f, 0.f, 0.f, 0.f, 0.f, 0.f};
#pragma unroll 4
    for (int ks = 0; ks < kH / 32; ks += 2) {
        acc0 = wmma_bf16(ld_frag(hRow, ks * 32, hiOff),
                         ld_frag(wRow, ks * 32, hiOff), acc0);
        acc1 = wmma_bf16(ld_frag(hRow, (ks + 1) * 32, hiOff),
                         ld_frag(wRow, (ks + 1) * 32, hiOff), acc1);
    }
    v8f acc = acc0 + acc1;
#pragma unroll
    for (int r = 0; r < 8; ++r) {
        const int m = mT * 16 + r + hiOff;
        y[(size_t)m * kO + nT * 16 + laneLow] = acc[r];
    }
}

// ---------------------------------------------------------------------------
extern "C" void kernel_launch(void* const* d_in, const int* in_sizes, int n_in,
                              void* d_out, int out_size, void* d_ws, size_t ws_size,
                              hipStream_t stream)
{
    const float* x   = (const float*)d_in[0];
    const float* bh  = (const float*)d_in[1];
    const float* Wix = (const float*)d_in[2];
    const float* Wih = (const float*)d_in[3];
    const float* Wox = (const float*)d_in[4];
    const float* Woh = (const float*)d_in[5];
    const float* Wfx = (const float*)d_in[6];
    const float* Wfh = (const float*)d_in[7];
    const float* Wzx = (const float*)d_in[8];
    const float* Wzh = (const float*)d_in[9];
    const float* bi  = (const float*)d_in[10];
    const float* bff = (const float*)d_in[11];
    const float* bo  = (const float*)d_in[12];
    const float* bz  = (const float*)d_in[13];
    const float* Wy  = (const float*)d_in[14];
    const float* by  = (const float*)d_in[15];

    // ---- workspace layout (bytes) ----
    char* ws = (char*)d_ws;
    size_t off = 0;
    __hip_bfloat16* xbf = (__hip_bfloat16*)(ws + off); off += (size_t)kB * kT * kI * 2;       // 16 MB
    __hip_bfloat16* wx[4];
    for (int g = 0; g < 4; ++g) { wx[g] = (__hip_bfloat16*)(ws + off); off += (size_t)kH * kI * 2; }
    __hip_bfloat16* wh[4];
    for (int g = 0; g < 4; ++g) { wh[g] = (__hip_bfloat16*)(ws + off); off += (size_t)kH * kH * 2; }
    __hip_bfloat16* wyb = (__hip_bfloat16*)(ws + off); off += (size_t)kO * kH * 2;
    __hip_bfloat16* hs_bf = (__hip_bfloat16*)(ws + off); off += (size_t)kB * kT * kH * 2;     // 64 MB
    float* cbuf = (float*)(ws + off); off += (size_t)kB * kH * 4;
    unsigned* bar = (unsigned*)(ws + off); off += 256;
    // total ~95.2 MB — must be <= ws_size

    float* y      = (float*)d_out;
    float* hs_f32 = (float*)d_out + (size_t)kB * kT * kO;

    // ---- 1) bf16 conversions ----
    auto conv = [&](const float* s, __hip_bfloat16* d, int n) {
        int blocks = (n + 255) / 256; if (blocks > 4096) blocks = 4096;
        f32_to_bf16_kernel<<<blocks, 256, 0, stream>>>(s, d, n);
    };
    conv(x,   xbf,   kB * kT * kI);
    conv(Wix, wx[0], kH * kI);  conv(Wox, wx[1], kH * kI);
    conv(Wfx, wx[2], kH * kI);  conv(Wzx, wx[3], kH * kI);
    conv(Wih, wh[0], kH * kH);  conv(Woh, wh[1], kH * kH);
    conv(Wfh, wh[2], kH * kH);  conv(Wzh, wh[3], kH * kH);
    conv(Wy,  wyb,   kO * kH);

    bar_init_kernel<<<1, 32, 0, stream>>>(bar);

    // ---- 2) persistent recurrent scan (gate order: i, o, f, z) ----
    lstm_scan_kernel<<<256, 128, 0, stream>>>(
        xbf,
        wx[0], wx[1], wx[2], wx[3],
        wh[0], wh[1], wh[2], wh[3],
        bi, bo, bff, bz,
        bh, hs_bf, cbuf, hs_f32, bar);

    // ---- 3) output projection ----
    out_proj_kernel<<<4096, 256, 0, stream>>>(hs_bf, wyb, by, y);
}
